// CrossVectorAttentionEncoder_22909355556982
// MI455X (gfx1250) — matrix operand order repaired and checked by reading, not compiled, not run on hardware
//
#include <hip/hip_runtime.h>
#include <stdint.h>

#define B_   2
#define NT_  1024
#define NS_  4096
#define DM_  512
#define H_   8
#define K_   16
#define DSP_ 2
#define DFF_ 2048
#define DH_  64
#define ROWS_ (B_*NT_)

typedef __attribute__((ext_vector_type(16))) _Float16 v16h;
typedef __attribute__((ext_vector_type(8)))  float    v8f;

// ---------------------------------------------------------------------------
// WMMA helpers (layouts per CDNA5 ISA 7.12.2, wave32)
// ---------------------------------------------------------------------------
static __device__ inline v8f wmma16(v16h a, v16h b, v8f c) {
  return __builtin_amdgcn_wmma_f32_16x16x32_f16(false, a, false, b, (short)0, c,
                                                false, false);
}

// A fragment: 16x32 f16 tile, row-major with leading dim lda (halfwords).
// Per-lane data is one contiguous 32B run (k = 16*half + j) -> two b128 loads.
// REQUIRES: lda % 8 == 0, kbase % 32 == 0 (=> 16B alignment).
static __device__ inline v16h ldA(const _Float16* A, int lda, int row0,
                                  int kbase, int lane) {
  int half = lane >> 4, m = lane & 15;
  const uint4* p = (const uint4*)(A + (row0 + m) * lda + kbase + 16 * half);
  union { uint4 u[2]; v16h v; } c;
  c.u[0] = p[0];   // halfwords 0..7
  c.u[1] = p[1];   // halfwords 8..15
  return c.v;
}

// B fragment from swizzled storage: matrix (Kd x N) stored as 32x16 tiles in
// fragment order, tile index = tn*(Kd/32)+tk, tile = [lane][16] halfwords:
//   value j of lane l = B[tk*32 + 16*(l>>4) + j][tn*16 + (l&15)]
static __device__ inline v16h ldB_sw(const _Float16* W, int tile, int lane) {
  const uint4* p = (const uint4*)(W + (size_t)tile * 512 + lane * 16);
  union { uint4 u[2]; v16h v; } c;
  c.u[0] = p[0];
  c.u[1] = p[1];
  return c.v;
}

// One 16x16 D tile: A[M x Kd] (row-major LDS) @ W (swizzled LDS)
static __device__ inline v8f tile_mm(const _Float16* A, int lda,
                                     const _Float16* W, int Kd32,
                                     int row0, int ntile, int lane) {
  v8f c;
#pragma unroll
  for (int r = 0; r < 8; ++r) c[r] = 0.f;
  for (int tk = 0; tk < Kd32; ++tk)
    c = wmma16(ldA(A, lda, row0, tk * 32, lane),
               ldB_sw(W, ntile * Kd32 + tk, lane), c);
  return c;
}

// ---------------------------------------------------------------------------
// conversion kernels
// ---------------------------------------------------------------------------
__global__ void k_cvt(const float* s, _Float16* d, int n) {
  int i = blockIdx.x * 256 + threadIdx.x;
  if (i < n) d[i] = (_Float16)s[i];
}

// f32 -> f16 with fragment swizzle (weight matrices, Kd x N row-major input)
__global__ void k_cvt_sw(const float* W, _Float16* dst, int Kd, int N) {
  int i = blockIdx.x * 256 + threadIdx.x;
  if (i >= Kd * N) return;
  int k = i / N, n = i - k * N;
  int tk = k >> 5, kr = k & 31, tn = n >> 4, nr = n & 15;
  int lane = ((kr >> 4) << 4) + nr, j = kr & 15;
  int Kd32 = Kd >> 5;
  dst[((size_t)(tn * Kd32 + tk) << 9) + lane * 16 + j] = (_Float16)W[i];
}

__global__ void k_rowsq(const float* src, float* s2) {
  int wave = threadIdx.x >> 5, lane = threadIdx.x & 31;
  int row = blockIdx.x * 8 + wave;   // over B*NS rows
  const float* p = src + (size_t)row * DM_;
  float s = 0.f;
  for (int j = lane; j < DM_; j += 32) { float v = p[j]; s += v * v; }
#pragma unroll
  for (int o = 16; o > 0; o >>= 1) s += __shfl_xor(s, o);
  if (lane == 0) s2[row] = s;
}

// ---------------------------------------------------------------------------
// scores[b,m,n] = |src_n|^2 - 2 * tgt_m . src_n   (ranking-equivalent to d2)
// 64(M) x 128(N) tile per block; B staged into LDS in swizzled order.
// ---------------------------------------------------------------------------
__global__ void __launch_bounds__(256)
k_scores(const _Float16* tgt16, const _Float16* src16, const float* s2,
         float* scores) {
  __shared__ uint4 As4[64 * 40 / 8];        // 64 x 40 halfwords (lda=40)
  __shared__ uint4 Bs4[8 * 64];             // 8 swizzled 32x16 tiles
  _Float16* As = (_Float16*)As4;
  _Float16* Bs = (_Float16*)Bs4;
  int tid = threadIdx.x, lane = tid & 31, wave = tid >> 5;
  int b = blockIdx.z, m0 = blockIdx.y * 64, n0 = blockIdx.x * 128;
  int wm = wave >> 1, wn = wave & 1;
  v8f acc[4];
  for (int t = 0; t < 4; ++t)
#pragma unroll
    for (int r = 0; r < 8; ++r) acc[t][r] = 0.f;

  // per-thread staging geometry (constant across k-loop)
  int ar = tid >> 2, aq = tid & 3;              // A: row, 8-halfword quad
  int bn_ = tid & 127, bkh = tid >> 7;          // B: col row, k half (0/1)
  int bdst = ((bn_ >> 4) << 6) + ((bkh << 4) + (bn_ & 15)) * 2;  // uint4 idx

  for (int k0 = 0; k0 < DM_; k0 += 32) {
    // A tile: one uint4 per thread (64x32 halfwords)
    As4[ar * 5 + aq] =
        *(const uint4*)(tgt16 + ((size_t)(b * NT_ + m0 + ar)) * DM_ + k0 + aq * 8);
    // B tile: 32B per thread, written swizzled (transpose via layout)
    {
      const uint4* g =
          (const uint4*)(src16 + ((size_t)(b * NS_ + n0 + bn_)) * DM_ + k0 + bkh * 16);
      Bs4[bdst]     = g[0];
      Bs4[bdst + 1] = g[1];
    }
    __syncthreads();
    v16h a = ldA(As, 40, wm * 16, 0, lane);
#pragma unroll
    for (int t = 0; t < 4; ++t)
      acc[t] = wmma16(a, ldB_sw(Bs, wn * 4 + t, lane), acc[t]);
    __syncthreads();
  }
  int half = lane >> 4, nn = lane & 15;
#pragma unroll
  for (int t = 0; t < 4; ++t) {
    int col = n0 + wn * 64 + t * 16 + nn;
    float s2v = s2[b * NS_ + col];
#pragma unroll
    for (int r = 0; r < 8; ++r) {
      int row = m0 + wm * 16 + r + 8 * half;
      scores[((size_t)(b * NT_ + row)) * NS_ + col] = s2v - 2.f * acc[t][r];
    }
  }
}

// ---------------------------------------------------------------------------
// top-K=16 smallest per row of NS=4096 (iterative argmin, ties -> lowest idx)
// ---------------------------------------------------------------------------
__global__ void k_topk(const float* scores, int* idx) {
  __shared__ float vals[NS_];
  __shared__ float rmin[256];
  __shared__ int   ridx[256];
  int tid = threadIdx.x;
  size_t row = blockIdx.x;
  const float* p = scores + row * NS_;
  for (int j = tid; j < NS_; j += 256) vals[j] = p[j];
  __syncthreads();
  for (int it = 0; it < K_; ++it) {
    float lm = 3.4e38f; int li = 0;
    for (int j = tid; j < NS_; j += 256) {
      float v = vals[j];
      if (v < lm) { lm = v; li = j; }
    }
    rmin[tid] = lm; ridx[tid] = li; __syncthreads();
    for (int s = 128; s > 0; s >>= 1) {
      if (tid < s) {
        float v2 = rmin[tid + s];
        if (v2 < rmin[tid] || (v2 == rmin[tid] && ridx[tid + s] < ridx[tid])) {
          rmin[tid] = v2; ridx[tid] = ridx[tid + s];
        }
      }
      __syncthreads();
    }
    if (tid == 0) { idx[row * K_ + it] = ridx[0]; vals[ridx[0]] = 3.4e38f; }
    __syncthreads();
  }
}

// ---------------------------------------------------------------------------
// Fused vector-attention kernel: one workgroup per (b,n).
// Rows flattened as (h*16+k) -> 128x64 activations; weights (swizzled,
// async-staged) + activations LDS-resident (~309KB of the 320KB WGP LDS).
// ---------------------------------------------------------------------------
#define OW_PSI 0          // 64x64   swizzled (Kd32=2,  4 ntiles)
#define OW_ALP 8192
#define OW_TH1 16384      // 64x256  swizzled (Kd32=2, 16 ntiles)
#define OW_TH2 49152      // 256x64  swizzled (Kd32=8,  4 ntiles)
#define OW_GA1 81920
#define OW_GA2 114688
#define OW_PHI 147456     // 64x64   swizzled
#define W_BYTES 155648
#define O_XS   155648     // 128x72 f16  (src gather, later alp+delta)
#define O_DIN  174080     // 128x72 f16  (tp-sp, later E_in)
#define O_HID  192512     // 128x264 f16 (MLP hidden)
#define O_DLT  260096     // 128x72 f16  (delta)
#define O_LGT  278528     // 128x64 f32  (logits)
#define O_PHIO 311296     // 8x64 f32
#define O_XPHI 313344     // 16x72 f16 (tgt head rows, zero-padded)
#define O_IDXS 315648     // 16 int
#define O_PK   315712     // 32 f32
#define ATTN_LDS_BYTES 315904

__global__ void __launch_bounds__(256)
k_attn(const float* tgt, const float* src, const float* tgt_pos,
       const float* src_pos, const int* idx, const _Float16* w16,
       const float* phi_b, const float* psi_b, const float* alpha_b,
       const float* th_b1, const float* th_b2, const float* ga_b1,
       const float* ga_b2, const float* up_w, const float* up_b,
       float* attn_out) {
  extern __shared__ char smem[];
  _Float16* Wpsi = (_Float16*)(smem + OW_PSI);
  _Float16* Walp = (_Float16*)(smem + OW_ALP);
  _Float16* Wth1 = (_Float16*)(smem + OW_TH1);
  _Float16* Wth2 = (_Float16*)(smem + OW_TH2);
  _Float16* Wga1 = (_Float16*)(smem + OW_GA1);
  _Float16* Wga2 = (_Float16*)(smem + OW_GA2);
  _Float16* Wphi = (_Float16*)(smem + OW_PHI);
  _Float16* Xs   = (_Float16*)(smem + O_XS);
  _Float16* Din  = (_Float16*)(smem + O_DIN);
  _Float16* Hid  = (_Float16*)(smem + O_HID);
  _Float16* Dlt  = (_Float16*)(smem + O_DLT);
  float* Lgt  = (float*)(smem + O_LGT);
  float* Phi  = (float*)(smem + O_PHIO);
  _Float16* Xphi = (_Float16*)(smem + O_XPHI);
  int*   idxs = (int*)(smem + O_IDXS);
  float* pk   = (float*)(smem + O_PK);

  int tid = threadIdx.x, lane = tid & 31, wave = tid >> 5;
  int bn = blockIdx.x, b = bn / NT_;

  // async-stage all weight matrices into LDS (152KB, already swizzled in ws).
  // GLOBAL_LOAD_ASYNC_TO_LDS_B128: 16B/lane, tracked by ASYNCcnt.
  {
    unsigned long long gbase = (unsigned long long)(const void*)w16 + tid * 16;
    unsigned ldsoff = tid * 16;                     // dynamic LDS starts at 0
    for (int i = 0; i < W_BYTES / (256 * 16); ++i) {
      unsigned long long ga = gbase + (unsigned)(i * 4096);
      unsigned lo = ldsoff + i * 4096;
      asm volatile("global_load_async_to_lds_b128 %0, %1, off"
                   :: "v"(lo), "v"(ga) : "memory");
    }
  }
  if (tid < K_) {
    int iv = idx[(size_t)bn * K_ + tid];
    idxs[tid] = iv;
    pk[2 * tid]     = src_pos[(size_t)(b * NS_ + iv) * DSP_ + 0];
    pk[2 * tid + 1] = src_pos[(size_t)(b * NS_ + iv) * DSP_ + 1];
  }
  __syncthreads();  // idxs/pk visible (async weights still in flight)

  // gather Xs[row=(h*16+k)][d] = src_sel ; Din = tgt_pos - sp (up_src proj)
  // float4 loads, b64 packed stores
  for (int e = tid; e < 128 * 16; e += 256) {
    int row = e >> 4, q = e & 15;
    int h = row >> 4, k = row & 15;
    int d0 = q * 4, dd = h * 64 + d0;
    int sidx = idxs[k];
    float4 sv = *(const float4*)(src + ((size_t)(b * NS_ + sidx)) * DM_ + dd);
    float4 tp = *(const float4*)(tgt_pos + (size_t)bn * DM_ + dd);
    float4 w0 = *(const float4*)(up_w + dd);
    float4 w1 = *(const float4*)(up_w + DM_ + dd);
    float4 ub = *(const float4*)(up_b + dd);
    float p0 = pk[2 * k], p1 = pk[2 * k + 1];
    union { _Float16 h4[4]; unsigned long long u; } xs, dn;
    xs.h4[0] = (_Float16)sv.x; xs.h4[1] = (_Float16)sv.y;
    xs.h4[2] = (_Float16)sv.z; xs.h4[3] = (_Float16)sv.w;
    dn.h4[0] = (_Float16)(tp.x - (p0 * w0.x + p1 * w1.x + ub.x));
    dn.h4[1] = (_Float16)(tp.y - (p0 * w0.y + p1 * w1.y + ub.y));
    dn.h4[2] = (_Float16)(tp.z - (p0 * w0.z + p1 * w1.z + ub.z));
    dn.h4[3] = (_Float16)(tp.w - (p0 * w0.w + p1 * w1.w + ub.w));
    *(unsigned long long*)(Xs + row * 72 + d0)  = xs.u;
    *(unsigned long long*)(Din + row * 72 + d0) = dn.u;
  }
  // stage tgt head-rows as a 16x64 tile (rows 8..15 zero) for the phi WMMA
  for (int e = tid; e < 16 * 64; e += 256) {
    int row = e >> 6, d = e & 63;
    Xphi[row * 72 + d] =
        (row < 8) ? (_Float16)tgt[(size_t)bn * DM_ + row * 64 + d] : (_Float16)0.f;
  }
  asm volatile("s_wait_asynccnt 0" ::: "memory");  // weights landed
  __syncthreads();

  int half = lane >> 4, nn = lane & 15;

  // phi = tgt_h @ phi_w + phi_b  (16x64 @ 64x64, rows 8..15 padding)
  if (wave < 4) {
    v8f c = tile_mm(Xphi, 72, Wphi, 2, 0, wave, lane);
    if (half == 0) {
      int col = wave * 16 + nn;
      float bia = phi_b[col];
#pragma unroll
      for (int r = 0; r < 8; ++r) Phi[r * 64 + col] = c[r] + bia;
    }
  }

  // Hid = relu(Din @ th_w1 + th_b1)   128x64 @ 64x256
  for (int t = wave; t < 128; t += 8) {
    int mt = t >> 4, nt = t & 15;
    v8f c = tile_mm(Din, 72, Wth1, 2, mt * 16, nt, lane);
    int col = nt * 16 + nn;
    float bia = th_b1[col];
#pragma unroll
    for (int r = 0; r < 8; ++r) {
      int row = mt * 16 + r + 8 * half;
      float v = c[r] + bia;
      Hid[row * 264 + col] = (_Float16)(v > 0.f ? v : 0.f);
    }
  }
  __syncthreads();
  // Dlt = Hid @ th_w2 + th_b2        128x256 @ 256x64
  for (int t = wave; t < 32; t += 8) {
    int mt = t >> 2, nt = t & 3;
    v8f c = tile_mm(Hid, 264, Wth2, 8, mt * 16, nt, lane);
    int col = nt * 16 + nn;
    float bia = th_b2[col];
#pragma unroll
    for (int r = 0; r < 8; ++r) {
      int row = mt * 16 + r + 8 * half;
      Dlt[row * 72 + col] = (_Float16)(c[r] + bia);
    }
  }
  __syncthreads();
  // psi/alp GEMMs from Xs; then Din <- phi - psi + delta ; Xs <- alp + delta
  {
    v8f cp[4], ca[4];
#pragma unroll
    for (int i = 0; i < 4; ++i) {
      int t = wave + 8 * i, mt = t >> 2, nt = t & 3;
      cp[i] = tile_mm(Xs, 72, Wpsi, 2, mt * 16, nt, lane);
      ca[i] = tile_mm(Xs, 72, Walp, 2, mt * 16, nt, lane);
    }
    __syncthreads();  // all reads of Xs done before overwrite
#pragma unroll
    for (int i = 0; i < 4; ++i) {
      int t = wave + 8 * i, mt = t >> 2, nt = t & 3;
      int col = nt * 16 + nn;
      float pb = psi_b[col], ab = alpha_b[col];
#pragma unroll
      for (int r = 0; r < 8; ++r) {
        int row = mt * 16 + r + 8 * half;
        int h = row >> 4;
        float dl = (float)Dlt[row * 72 + col];
        Din[row * 72 + col] = (_Float16)(Phi[h * 64 + col] - (cp[i][r] + pb) + dl);
        Xs[row * 72 + col]  = (_Float16)(ca[i][r] + ab + dl);
      }
    }
  }
  __syncthreads();
  // Hid = relu(Din @ ga_w1 + ga_b1)
  for (int t = wave; t < 128; t += 8) {
    int mt = t >> 4, nt = t & 15;
    v8f c = tile_mm(Din, 72, Wga1, 2, mt * 16, nt, lane);
    int col = nt * 16 + nn;
    float bia = ga_b1[col];
#pragma unroll
    for (int r = 0; r < 8; ++r) {
      int row = mt * 16 + r + 8 * half;
      float v = c[r] + bia;
      Hid[row * 264 + col] = (_Float16)(v > 0.f ? v : 0.f);
    }
  }
  __syncthreads();
  // Lgt = Hid @ ga_w2 + ga_b2  (f32)
  for (int t = wave; t < 32; t += 8) {
    int mt = t >> 2, nt = t & 3;
    v8f c = tile_mm(Hid, 264, Wga2, 8, mt * 16, nt, lane);
    int col = nt * 16 + nn;
    float bia = ga_b2[col];
#pragma unroll
    for (int r = 0; r < 8; ++r) {
      int row = mt * 16 + r + 8 * half;
      Lgt[row * 64 + col] = c[r] + bia;
    }
  }
  __syncthreads();
  // per-(h,d) softmax over K, weighted sum with (alp+delta) in Xs
  for (int e = tid; e < H_ * DH_; e += 256) {
    int h = e >> 6, d = e & 63;
    float mx = -3.4e38f;
#pragma unroll
    for (int k = 0; k < K_; ++k) mx = fmaxf(mx, Lgt[(h * 16 + k) * 64 + d]);
    float s = 0.f, o = 0.f;
#pragma unroll
    for (int k = 0; k < K_; ++k) {
      float w = __expf(Lgt[(h * 16 + k) * 64 + d] - mx);
      s += w;
      o += w * (float)Xs[(h * 16 + k) * 72 + d];
    }
    attn_out[(size_t)bn * DM_ + h * 64 + d] = o / s;
  }
}

// ---------------------------------------------------------------------------
// out = LayerNorm(a + r) * w + bi ; optional f16 mirror (for next GEMM)
// ---------------------------------------------------------------------------
__global__ void k_ln(const float* a, const float* r, const float* w,
                     const float* bi, float* out32, _Float16* out16) {
  __shared__ float red[256];
  int tid = threadIdx.x;
  size_t row = blockIdx.x;
  float v0 = a[row * DM_ + tid]       + r[row * DM_ + tid];
  float v1 = a[row * DM_ + tid + 256] + r[row * DM_ + tid + 256];
  red[tid] = v0 + v1; __syncthreads();
  for (int s = 128; s > 0; s >>= 1) { if (tid < s) red[tid] += red[tid + s]; __syncthreads(); }
  float mean = red[0] / (float)DM_; __syncthreads();
  float d0 = v0 - mean, d1 = v1 - mean;
  red[tid] = d0 * d0 + d1 * d1; __syncthreads();
  for (int s = 128; s > 0; s >>= 1) { if (tid < s) red[tid] += red[tid + s]; __syncthreads(); }
  float rs = rsqrtf(red[0] / (float)DM_ + 1e-5f);
  float o0 = d0 * rs * w[tid] + bi[tid];
  float o1 = d1 * rs * w[tid + 256] + bi[tid + 256];
  out32[row * DM_ + tid] = o0;
  out32[row * DM_ + tid + 256] = o1;
  if (out16) {
    out16[row * DM_ + tid] = (_Float16)o0;
    out16[row * DM_ + tid + 256] = (_Float16)o1;
  }
}

// ---------------------------------------------------------------------------
// Generic WMMA GEMM: C[MxN] = A[MxKd] @ Bsw[KdxN, swizzled] + bias
// mode 0: relu -> f16 out ; mode 1: f32 out
// ---------------------------------------------------------------------------
__global__ void __launch_bounds__(256)
k_gemm(const _Float16* A, const _Float16* Bw, const float* bias, void* Cout,
       int M, int N, int Kd, int mode) {
  __shared__ uint4 As4[64 * 40 / 8];  // 64 x 40 halfwords (lda=40)
  __shared__ uint4 Bs4[8 * 64];       // 8 swizzled 32x16 tiles (1KB each)
  _Float16* As = (_Float16*)As4;
  _Float16* Bs = (_Float16*)Bs4;
  int tid = threadIdx.x, lane = tid & 31, wave = tid >> 5;
  int m0 = blockIdx.y * 64, n0 = blockIdx.x * 128;
  int wm = wave >> 1, wn = wave & 1;
  int Kd32 = Kd >> 5, tn0 = n0 >> 4;
  const uint4* Bw4 = (const uint4*)Bw;
  int ar = tid >> 2, aq = tid & 3;    // A staging: row, 8-halfword quad
  v8f acc[4];
  for (int t = 0; t < 4; ++t)
#pragma unroll
    for (int r = 0; r < 8; ++r) acc[t][r] = 0.f;

  for (int k0 = 0; k0 < Kd; k0 += 32) {
    int tk = k0 >> 5;
    // A tile: one uint4 per thread
    As4[ar * 5 + aq] =
        *(const uint4*)(A + (size_t)(m0 + ar) * Kd + k0 + aq * 8);
    // copy 8 contiguous swizzled B tiles (64 uint4 each)
#pragma unroll
    for (int i = 0; i < 2; ++i) {
      int e = tid + i * 256, t = e >> 6, w = e & 63;
      Bs4[e] = Bw4[(size_t)((tn0 + t) * Kd32 + tk) * 64 + w];
    }
    __syncthreads();
    v16h a = ldA(As, 40, wm * 16, 0, lane);
#pragma unroll
    for (int t = 0; t < 4; ++t)
      acc[t] = wmma16(a, ldB_sw(Bs, wn * 4 + t, lane), acc[t]);
    __syncthreads();
  }
  int half = lane >> 4, nn = lane & 15;
#pragma unroll
  for (int t = 0; t < 4; ++t) {
    int col = n0 + wn * 64 + t * 16 + nn;
    float bia = bias[col];
#pragma unroll
    for (int r = 0; r < 8; ++r) {
      int row = m0 + wm * 16 + r + 8 * half;
      float v = acc[t][r] + bia;
      if (mode == 0)
        ((_Float16*)Cout)[(size_t)row * N + col] = (_Float16)(v > 0.f ? v : 0.f);
      else
        ((float*)Cout)[(size_t)row * N + col] = v;
    }
  }
}

// ---------------------------------------------------------------------------
extern "C" void kernel_launch(void* const* d_in, const int* in_sizes, int n_in,
                              void* d_out, int out_size, void* d_ws,
                              size_t ws_size, hipStream_t stream) {
  const float* tgt      = (const float*)d_in[0];
  const float* src      = (const float*)d_in[1];
  const float* tgt_pos  = (const float*)d_in[2];
  const float* src_pos  = (const float*)d_in[3];
  const float* phi_w    = (const float*)d_in[4];
  const float* phi_b    = (const float*)d_in[5];
  const float* psi_w    = (const float*)d_in[6];
  const float* psi_b    = (const float*)d_in[7];
  const float* alpha_w  = (const float*)d_in[8];
  const float* alpha_b  = (const float*)d_in[9];
  const float* th_w1    = (const float*)d_in[10];
  const float* th_b1    = (const float*)d_in[11];
  const float* th_w2    = (const float*)d_in[12];
  const float* th_b2    = (const float*)d_in[13];
  const float* ga_w1    = (const float*)d_in[14];
  const float* ga_b1    = (const float*)d_in[15];
  const float* ga_w2    = (const float*)d_in[16];
  const float* ga_b2    = (const float*)d_in[17];
  const float* up_w     = (const float*)d_in[18];
  const float* up_b     = (const float*)d_in[19];
  const float* ln_w     = (const float*)d_in[20];
  const float* ln_b     = (const float*)d_in[21];
  const float* ffn_w1   = (const float*)d_in[22];
  const float* ffn_b1   = (const float*)d_in[23];
  const float* ffn_w2   = (const float*)d_in[24];
  const float* ffn_b2   = (const float*)d_in[25];
  const float* ffn_ln_w = (const float*)d_in[26];
  const float* ffn_ln_b = (const float*)d_in[27];
  (void)in_sizes; (void)n_in; (void)out_size; (void)ws_size;

  char* ws = (char*)d_ws;
  size_t off = 0;
  auto alloc = [&](size_t bytes) -> char* {
    char* p = ws + off;
    off += (bytes + 255) & ~(size_t)255;
    return p;
  };
  float*     scores = (float*)alloc((size_t)B_ * NT_ * NS_ * 4);   // 32 MB
  int*       idxb   = (int*)alloc((size_t)B_ * NT_ * K_ * 4);
  float*     s2     = (float*)alloc((size_t)B_ * NS_ * 4);
  _Float16*  tgt16  = (_Float16*)alloc((size_t)B_ * NT_ * DM_ * 2);
  _Float16*  src16  = (_Float16*)alloc((size_t)B_ * NS_ * DM_ * 2);
  _Float16*  w16    = (_Float16*)alloc(W_BYTES);         // 7 mats, swizzled
  _Float16*  ff1_16 = (_Float16*)alloc((size_t)DM_ * DFF_ * 2);   // swizzled
  _Float16*  ff2_16 = (_Float16*)alloc((size_t)DFF_ * DM_ * 2);   // swizzled
  float*     attn   = (float*)alloc((size_t)ROWS_ * DM_ * 4);
  float*     x      = (float*)alloc((size_t)ROWS_ * DM_ * 4);
  _Float16*  x16    = (_Float16*)alloc((size_t)ROWS_ * DM_ * 2);
  _Float16*  hid16  = (_Float16*)alloc((size_t)ROWS_ * DFF_ * 2);
  float*     y      = (float*)alloc((size_t)ROWS_ * DM_ * 4);

  auto cvt = [&](const float* s, _Float16* d, int n) {
    k_cvt<<<(n + 255) / 256, 256, 0, stream>>>(s, d, n);
  };
  auto cvt_sw = [&](const float* s, _Float16* d, int Kd, int N) {
    k_cvt_sw<<<(Kd * N + 255) / 256, 256, 0, stream>>>(s, d, Kd, N);
  };
  cvt(tgt, tgt16, B_ * NT_ * DM_);
  cvt(src, src16, B_ * NS_ * DM_);
  cvt_sw(psi_w,   (_Float16*)((char*)w16 + OW_PSI), DH_, DH_);
  cvt_sw(alpha_w, (_Float16*)((char*)w16 + OW_ALP), DH_, DH_);
  cvt_sw(th_w1,   (_Float16*)((char*)w16 + OW_TH1), DH_, 4 * DH_);
  cvt_sw(th_w2,   (_Float16*)((char*)w16 + OW_TH2), 4 * DH_, DH_);
  cvt_sw(ga_w1,   (_Float16*)((char*)w16 + OW_GA1), DH_, 4 * DH_);
  cvt_sw(ga_w2,   (_Float16*)((char*)w16 + OW_GA2), 4 * DH_, DH_);
  cvt_sw(phi_w,   (_Float16*)((char*)w16 + OW_PHI), DH_, DH_);
  cvt_sw(ffn_w1, ff1_16, DM_, DFF_);
  cvt_sw(ffn_w2, ff2_16, DFF_, DM_);

  k_rowsq<<<(B_ * NS_) / 8, 256, 0, stream>>>(src, s2);
  k_scores<<<dim3(NS_ / 128, NT_ / 64, B_), 256, 0, stream>>>(tgt16, src16, s2,
                                                              scores);
  k_topk<<<B_ * NT_, 256, 0, stream>>>(scores, idxb);
  k_attn<<<B_ * NT_, 256, ATTN_LDS_BYTES, stream>>>(
      tgt, src, tgt_pos, src_pos, idxb, w16, phi_b, psi_b, alpha_b,
      th_b1, th_b2, ga_b1, ga_b2, up_w, up_b, attn);
  k_ln<<<B_ * NT_, 256, 0, stream>>>(tgt, attn, ln_w, ln_b, x, x16);
  k_gemm<<<dim3(DFF_ / 128, ROWS_ / 64), 256, 0, stream>>>(
      x16, ff1_16, ffn_b1, hid16, ROWS_, DFF_, DM_, 0);
  k_gemm<<<dim3(DM_ / 128, ROWS_ / 64), 256, 0, stream>>>(
      hid16, ff2_16, ffn_b2, y, ROWS_, DM_, DFF_, 1);
  k_ln<<<B_ * NT_, 256, 0, stream>>>(x, y, ffn_ln_w, ffn_ln_b, (float*)d_out,
                                     nullptr);
}
